// STAttentionBlock_63136019251493
// MI455X (gfx1250) — matrix operand order repaired and checked by reading, not compiled, not run on hardware
//
#include <hip/hip_runtime.h>
#include <math.h>

typedef __attribute__((ext_vector_type(16))) _Float16 v16h;
typedef __attribute__((ext_vector_type(8)))  _Float16 v8h;
typedef __attribute__((ext_vector_type(8)))  float    v8f;

#define DEV static __device__ __forceinline__

DEV float gelu_f(float x) { return 0.5f * x * (1.0f + erff(x * 0.70710678118654752f)); }

// A-fragment (16-bit A 16x32, ISA 7.12.2): lanes 0-15 -> K 0..7 | 16..23 ; lanes 16-31 -> K 8..15 | 24..31
DEV v16h fragA_from_lds(const _Float16* rowBase, int hiHalf) {
  const v8h x0 = *(const v8h*)(rowBase + (hiHalf ? 8 : 0));
  const v8h x1 = *(const v8h*)(rowBase + (hiHalf ? 24 : 16));
  v16h r;
#pragma unroll
  for (int i = 0; i < 8; ++i) { r[i] = x0[i]; r[i + 8] = x1[i]; }
  return r;
}

// B-fragment (16-bit B 32x16): lanes 0-15 -> K 0..7 | 16..23 ; lanes 16-31 -> K 8..15 | 24..31 (matches A)
DEV v16h fragB_from_lds(const _Float16* colBase, int hiHalf) {
  const v8h x0 = *(const v8h*)(colBase + hiHalf * 8);
  const v8h x1 = *(const v8h*)(colBase + 16 + hiHalf * 8);
  v16h r;
#pragma unroll
  for (int i = 0; i < 8; ++i) { r[i] = x0[i]; r[i + 8] = x1[i]; }
  return r;
}

union HPack { _Float16 h[4]; uint2 u; };

// ---------------------------------------------------------------------------
// Generic batched WMMA GEMM:  C[b] = A[b](MxK) @ B[b](KxN) + bias
// Block tile 64(M) x 32(N); 4 waves, each wave 32x16 via two v_wmma_f32_16x16x32_f16.
// Double-buffered f16 LDS; phase-split loads (all global loads clause together,
// then convert+ds_store); fragment ds_loads issued before next-tile stores so the
// WMMA only waits on its own loads.
// Requirements (all call sites satisfy): M%64==0, N%32==0, K%32==0, lda%4==0.
// Output column j -> (j>>colShift)*colStrideOuter + (j & ((1<<colShift)-1))*colStrideInner.
// ---------------------------------------------------------------------------
__global__ __launch_bounds__(128)
void gemm_wmma_kernel(const float* __restrict__ A, const float* __restrict__ B,
                      const float* __restrict__ bias, float* __restrict__ C,
                      int M, int N, int K,
                      long long sAb, long long sBb, long long sCb,
                      int lda, int ldb,
                      long long cRow, int colShift, int colStrideInner,
                      long long colStrideOuter) {
  __shared__ __align__(16) _Float16 sA[2][64 * 32];   // [m][k]
  __shared__ __align__(16) _Float16 sB[2][32 * 32];   // [n][k]

  const int b  = blockIdx.z;
  const int m0 = blockIdx.y * 64;
  const int n0 = blockIdx.x * 32;
  const float* Ab = A + (long long)b * sAb;
  const float* Bb = B + (long long)b * sBb;
  float*       Cb = C + (long long)b * sCb;

  const int tid  = threadIdx.x;
  const int wave = tid >> 5;
  const int lane = tid & 31;
  const int h    = lane & 15;
  const int hiH  = lane >> 4;
  const int wm   = (wave >> 1) * 32;   // wave's M offset within block tile
  const int wn   = (wave & 1) * 16;    // wave's N offset within block tile

  auto load_tiles = [&](int k0, _Float16* dA, _Float16* dB) {
    // phase 1: issue ALL global loads (clause-able, one wait)
    float4 fa[4];
#pragma unroll
    for (int i = 0; i < 4; ++i) {
      int e  = tid + i * 128;          // float4 index, 512 total
      int r  = e >> 3;                 // 8 float4 per row
      int c4 = (e & 7) * 4;
      fa[i] = *(const float4*)(Ab + (long long)(m0 + r) * lda + (k0 + c4));
    }
    float fb[2][4];
#pragma unroll
    for (int i = 0; i < 2; ++i) {
      int e  = tid + i * 128;          // 256 tasks
      int n  = e & 31;
      int kg = (e >> 5) * 4;
      const float* bp = Bb + (long long)(k0 + kg) * ldb + (n0 + n);
#pragma unroll
      for (int j = 0; j < 4; ++j) fb[i][j] = bp[j * ldb];
    }
    // phase 2: convert and store to LDS
#pragma unroll
    for (int i = 0; i < 4; ++i) {
      int e  = tid + i * 128;
      int r  = e >> 3;
      int c4 = (e & 7) * 4;
      HPack p;
      p.h[0] = (_Float16)fa[i].x; p.h[1] = (_Float16)fa[i].y;
      p.h[2] = (_Float16)fa[i].z; p.h[3] = (_Float16)fa[i].w;
      *(uint2*)&dA[r * 32 + c4] = p.u;
    }
#pragma unroll
    for (int i = 0; i < 2; ++i) {
      int e  = tid + i * 128;
      int n  = e & 31;
      int kg = (e >> 5) * 4;
      HPack p;
      p.h[0] = (_Float16)fb[i][0]; p.h[1] = (_Float16)fb[i][1];
      p.h[2] = (_Float16)fb[i][2]; p.h[3] = (_Float16)fb[i][3];
      *(uint2*)&dB[n * 32 + kg] = p.u;
    }
  };

  v8f acc0 = {}; v8f acc1 = {};

  load_tiles(0, sA[0], sB[0]);
  __syncthreads();

  int buf = 0;
  for (int k0 = 0; k0 < K; k0 += 32) {
    // fragment loads for current buffer FIRST (LDS ops retire in order: the
    // WMMA then only waits for these, not for the next tile's stores)
    v16h bf = fragB_from_lds(&sB[buf][(wn + h) * 32], hiH);
    v16h a0 = fragA_from_lds(&sA[buf][(wm + h) * 32], hiH);
    v16h a1 = fragA_from_lds(&sA[buf][(wm + 16 + h) * 32], hiH);
    if (k0 + 32 < K) {
      __builtin_prefetch(Ab + (long long)(m0 + (tid & 63)) * lda + (k0 + 64), 0, 1);
      load_tiles(k0 + 32, sA[buf ^ 1], sB[buf ^ 1]);
    }
    acc0 = __builtin_amdgcn_wmma_f32_16x16x32_f16(false, a0, false, bf, (short)0, acc0, false, false);
    acc1 = __builtin_amdgcn_wmma_f32_16x16x32_f16(false, a1, false, bf, (short)0, acc1, false, false);
    __syncthreads();
    buf ^= 1;
  }

  // C/D layout: lanes 0-15: n=lane, m=vgpr ; lanes 16-31: n=lane-16, m=vgpr+8
  const int gn = n0 + wn + h;
  const float bv = bias ? bias[gn] : 0.0f;
  const long long coff = (long long)(gn >> colShift) * colStrideOuter +
                         (long long)(gn & ((1 << colShift) - 1)) * colStrideInner;
#pragma unroll
  for (int r = 0; r < 8; ++r) {
    const int gm = m0 + wm + hiH * 8 + r;
    Cb[(long long)gm * cRow + coff]        = acc0[r] + bv;
    Cb[(long long)(gm + 16) * cRow + coff] = acc1[r] + bv;
  }
}

// ---------------------------------------------------------------------------
// Spatial attention scores: att[n,u,v,s] = tanh(sum_{t,c} q*k /1024)*alpha[s] + att0[u,v,s]
// qk layout (N,V,T,IC,6): q = ch 0..2, k = ch 3..5
// ---------------------------------------------------------------------------
__global__ void satt_kernel(const float* __restrict__ qk, const float* __restrict__ alphas,
                            const float* __restrict__ att0, float* __restrict__ att, int Nn) {
  int idx = blockIdx.x * blockDim.x + threadIdx.x;
  int total = Nn * 25 * 25 * 3;
  if (idx >= total) return;
  int s = idx % 3, v = (idx / 3) % 25, u = (idx / 75) % 25, n = idx / 1875;
  const float* qb = qk + (long long)(n * 25 + u) * 64 * 96 + s;
  const float* kb = qk + (long long)(n * 25 + v) * 64 * 96 + 3 + s;
  float acc = 0.f;
  for (int t = 0; t < 64; ++t) {
    const float* qr = qb + t * 96;
    const float* kr = kb + t * 96;
#pragma unroll
    for (int c = 0; c < 16; ++c) acc += qr[c * 6] * kr[c * 6];
  }
  att[idx] = tanhf(acc * (1.0f / 1024.0f)) * alphas[s] + att0[(u * 25 + v) * 3 + s];
}

// y192[(n,v,t)][c*3+s] = sum_u x[n,u,t,c] * att[n,u,v,s]
__global__ void smix_kernel(const float* __restrict__ x, const float* __restrict__ att,
                            float* __restrict__ y, int Nn) {
  long long idx = (long long)blockIdx.x * blockDim.x + threadIdx.x;
  long long total = (long long)Nn * 25 * 64 * 192;
  if (idx >= total) return;
  int cs = (int)(idx % 192); int c = cs / 3, s = cs - c * 3;
  int t = (int)((idx / 192) % 64);
  int v = (int)((idx / (192 * 64)) % 25);
  int n = (int)(idx / (192LL * 64 * 25));
  const float* xb = x + (long long)n * 25 * 4096 + t * 64 + c;      // + u*4096
  const float* ab = att + (long long)n * 1875 + v * 3 + s;          // + u*75
  float acc = 0.f;
#pragma unroll 5
  for (int u = 0; u < 25; ++u) acc += xb[(long long)u * 4096] * ab[u * 75];
  y[idx] = acc;
}

// LayerNorm over last-two-dims: one block per group of L contiguous elements, in-place.
__global__ void ln2_kernel(float* __restrict__ d, const float* __restrict__ g,
                           const float* __restrict__ bta, int L) {
  float* p = d + (long long)blockIdx.x * L;
  __shared__ float red[256];
  float s = 0.f, s2 = 0.f;
  for (int i = threadIdx.x; i < L; i += 256) { float v = p[i]; s += v; s2 += v * v; }
  red[threadIdx.x] = s; __syncthreads();
  for (int o = 128; o > 0; o >>= 1) { if (threadIdx.x < o) red[threadIdx.x] += red[threadIdx.x + o]; __syncthreads(); }
  float mean = red[0] / (float)L; __syncthreads();
  red[threadIdx.x] = s2; __syncthreads();
  for (int o = 128; o > 0; o >>= 1) { if (threadIdx.x < o) red[threadIdx.x] += red[threadIdx.x + o]; __syncthreads(); }
  float var = red[0] / (float)L - mean * mean;
  float rinv = rsqrtf(var + 1e-5f);
  for (int i = threadIdx.x; i < L; i += 256) p[i] = (p[i] - mean) * rinv * g[i] + bta[i];
}

__global__ void add_gelu_kernel(const float* __restrict__ a, const float* __restrict__ b,
                                float* __restrict__ o, long long total) {
  long long i = (long long)blockIdx.x * blockDim.x + threadIdx.x;
  if (i < total) o[i] = gelu_f(a[i] + b[i]);
}

// (N,V,T,C) -> (N,T,V,C)
__global__ void transpose_vt_kernel(const float* __restrict__ src, float* __restrict__ dst, int Nn) {
  long long i = (long long)blockIdx.x * blockDim.x + threadIdx.x;
  long long total = (long long)Nn * 25 * 64 * 64;
  if (i >= total) return;
  int c = (int)(i % 64);
  int t = (int)((i / 64) % 64);
  int v = (int)((i / 4096) % 25);
  int n = (int)(i / 102400);
  dst[(((long long)n * 64 + t) * 25 + v) * 64 + c] = src[i];
}

// Temporal scores, stored transposed for the mix GEMM: att[(n*4+ds)*64+q][t]
// qkt (N,T,V,IC,8): q channels ds (0..3), k channels ds+4. Masks: f(ds<2): t>=q ; b: t<=q.
__global__ void tatt_kernel(const float* __restrict__ qkt, const float* __restrict__ af,
                            const float* __restrict__ ab, float* __restrict__ att, int Nn) {
  int idx = blockIdx.x * blockDim.x + threadIdx.x;
  int total = Nn * 4 * 64 * 64;
  if (idx >= total) return;
  int t = idx % 64, q = (idx / 64) % 64, ds = (idx / 4096) % 4, n = idx / 16384;
  const float* qb = qkt + ((long long)(n * 64 + t) * 25) * 128 + ds;
  const float* kb = qkt + ((long long)(n * 64 + q) * 25) * 128 + ds + 4;
  float acc = 0.f;
  for (int v = 0; v < 25; ++v) {
    const float* qr = qb + v * 128;
    const float* kr = kb + v * 128;
#pragma unroll
    for (int c = 0; c < 16; ++c) acc += qr[c * 8] * kr[c * 8];
  }
  float alpha = (ds < 2) ? af[ds] : ab[ds - 2];
  float val = tanhf(acc * (1.0f / 400.0f)) * alpha;
  bool keep = (ds < 2) ? (t >= q) : (t <= q);
  att[idx] = keep ? val : 0.0f;
}

// conv (7,1) over T, z (N,T,V,C) channel-last view of zc (N,C,T,V); out (N,T,V,Cout)
__global__ void conv_kernel(const float* __restrict__ z, const float* __restrict__ w,
                            const float* __restrict__ bias, float* __restrict__ out, int Nn) {
  long long idx = (long long)blockIdx.x * blockDim.x + threadIdx.x;
  long long total = (long long)Nn * 64 * 25 * 64;
  if (idx >= total) return;
  int co = (int)(idx % 64);
  int v  = (int)((idx / 64) % 25);
  int t  = (int)((idx / 1600) % 64);
  int n  = (int)(idx / 102400);
  float acc = bias[co];
  for (int kh = 0; kh < 7; ++kh) {
    int tt = t + kh - 3;
    if (tt < 0 || tt >= 64) continue;
    const float* zr = z + (((long long)n * 64 + tt) * 25 + v) * 64;
    const float* wr = w + (long long)co * 448 + kh;
#pragma unroll 16
    for (int ci = 0; ci < 64; ++ci) acc += zr[ci] * wr[ci * 7];
  }
  out[idx] = acc;
}

__global__ void zero_kernel(float* p, int n) {
  int i = blockIdx.x * blockDim.x + threadIdx.x;
  if (i < n) p[i] = 0.f;
}

// per-channel sum/sumsq over rows of 64 channels -> stats[0..63]=sum, [64..127]=sumsq
__global__ void bnstats_kernel(const float* __restrict__ co, float* __restrict__ stats,
                               long long rows) {
  int ch  = threadIdx.x & 63;
  int sub = threadIdx.x >> 6;   // 0..3
  float s = 0.f, s2 = 0.f;
  for (long long r = (long long)blockIdx.x * 4 + sub; r < rows; r += (long long)gridDim.x * 4) {
    float v = co[r * 64 + ch];
    s += v; s2 += v * v;
  }
  __shared__ float sh[2][4][64];
  sh[0][sub][ch] = s; sh[1][sub][ch] = s2;
  __syncthreads();
  if (sub == 0) {
    float a = sh[0][0][ch] + sh[0][1][ch] + sh[0][2][ch] + sh[0][3][ch];
    float b = sh[1][0][ch] + sh[1][1][ch] + sh[1][2][ch] + sh[1][3][ch];
    atomicAdd(&stats[ch], a);
    atomicAdd(&stats[64 + ch], b);
  }
}

// out[n,v,t,c] = gelu( z4[n,t,v,c] + BN(conv[n,t,v,c]) )
__global__ void final_kernel(const float* __restrict__ z4, const float* __restrict__ co,
                             const float* __restrict__ stats, const float* __restrict__ g,
                             const float* __restrict__ b, float* __restrict__ out, int Nn) {
  long long idx = (long long)blockIdx.x * blockDim.x + threadIdx.x;
  long long total = (long long)Nn * 25 * 64 * 64;
  if (idx >= total) return;
  int c = (int)(idx % 64);
  int t = (int)((idx / 64) % 64);
  int v = (int)((idx / 4096) % 25);
  int n = (int)(idx / 102400);
  long long src = (((long long)n * 64 + t) * 25 + v) * 64 + c;
  float M = (float)((long long)Nn * 64 * 25);
  float mean = stats[c] / M;
  float var  = stats[64 + c] / M - mean * mean;
  float z1 = (co[src] - mean) * rsqrtf(var + 1e-5f) * g[c] + b[c];
  out[idx] = gelu_f(z4[src] + z1);
}

// ---------------------------------------------------------------------------
extern "C" void kernel_launch(void* const* d_in, const int* in_sizes, int n_in,
                              void* d_out, int out_size, void* d_ws, size_t ws_size,
                              hipStream_t stream) {
  const float* x        = (const float*)d_in[0];
  const float* w_in_s   = (const float*)d_in[1];
  const float* b_in_s   = (const float*)d_in[2];
  const float* alphas   = (const float*)d_in[3];
  const float* att0s    = (const float*)d_in[4];
  const float* w_out_s  = (const float*)d_in[5];
  const float* b_out_s  = (const float*)d_in[6];
  const float* lnos_g   = (const float*)d_in[7];
  const float* lnos_b   = (const float*)d_in[8];
  const float* w_ff_s   = (const float*)d_in[9];
  const float* b_ff_s   = (const float*)d_in[10];
  const float* lnfs_g   = (const float*)d_in[11];
  const float* lnfs_b   = (const float*)d_in[12];
  const float* w_in_t   = (const float*)d_in[13];
  const float* b_in_t   = (const float*)d_in[14];
  const float* alphat_f = (const float*)d_in[15];
  const float* alphat_b = (const float*)d_in[16];
  const float* w_out_t  = (const float*)d_in[17];
  const float* b_out_t  = (const float*)d_in[18];
  const float* lnot_g   = (const float*)d_in[19];
  const float* lnot_b   = (const float*)d_in[20];
  const float* w_ff_t   = (const float*)d_in[21];
  const float* b_ff_t   = (const float*)d_in[22];
  const float* lnft_g   = (const float*)d_in[23];
  const float* lnft_b   = (const float*)d_in[24];
  const float* conv_w   = (const float*)d_in[25];
  const float* conv_b   = (const float*)d_in[26];
  const float* bn_g     = (const float*)d_in[27];
  const float* bn_b     = (const float*)d_in[28];
  float* out = (float*)d_out;

  const int Nn = in_sizes[0] / (25 * 64 * 64);        // 128
  const int R  = Nn * 25 * 64;                        // 204800 rows

  // workspace layout (floats), regions reused across dead ranges
  float* W = (float*)d_ws;
  float* QK   = W;                                    // Nn*1600*96   | later QKT Nn*1600*128
  float* Y192 = W + (long long)Nn * 204800;           // Nn*1600*192  | later Z256 Nn*1600*256
  float* R1   = W + (long long)Nn * 614400;           // B1 / B3 / Zo / Z3
  float* R2   = W + (long long)Nn * 716800;           // B2 / Z2 / CO
  float* R3   = W + (long long)Nn * 819200;           // B4 / Z4
  float* Yt   = W + (long long)Nn * 921600;           // transposed activations
  float* ATTS = W + (long long)Nn * 1024000;          // Nn*1875
  float* ATTT = W + (long long)Nn * 1026048;          // Nn*16384
  float* STATS = W + (long long)Nn * 1042432;         // 128
  float* QKT  = QK;
  float* Z256 = Y192;
  (void)n_in; (void)out_size; (void)ws_size;

  auto gemm = [&](const float* A, const float* B, const float* bias, float* C,
                  int M, int N, int K, int batch,
                  long long sAb, long long sBb, long long sCb,
                  int lda, int ldb, long long cRow,
                  int colShift, int colStrideInner, long long colStrideOuter) {
    dim3 g((unsigned)(N / 32), (unsigned)(M / 64), (unsigned)batch);
    gemm_wmma_kernel<<<g, 128, 0, stream>>>(A, B, bias, C, M, N, K, sAb, sBb, sCb,
                                            lda, ldb, cRow, colShift, colStrideInner,
                                            colStrideOuter);
  };
  const long long E = (long long)Nn * 102400;         // elements per (N,*,*,C) tensor
  const int EB = (int)((E + 255) / 256);

  // ---- spatial branch ----
  gemm(x, w_in_s, b_in_s, QK, R, 96, 64, 1, 0, 0, 0, 64, 96, 96, 30, 1, 0);
  satt_kernel<<<(Nn * 1875 + 255) / 256, 256, 0, stream>>>(QK, alphas, att0s, ATTS, Nn);
  smix_kernel<<<(int)(((long long)Nn * 307200 + 255) / 256), 256, 0, stream>>>(x, ATTS, Y192, Nn);
  gemm(Y192, w_out_s, b_out_s, R1, R, 64, 192, 1, 0, 0, 0, 192, 64, 64, 30, 1, 0);
  ln2_kernel<<<Nn * 25, 256, 0, stream>>>(R1, lnos_g, lnos_b, 4096);
  add_gelu_kernel<<<EB, 256, 0, stream>>>(x, R1, R2, E);
  gemm(R2, w_ff_s, b_ff_s, R1, R, 64, 64, 1, 0, 0, 0, 64, 64, 64, 30, 1, 0);
  ln2_kernel<<<Nn * 25, 256, 0, stream>>>(R1, lnfs_g, lnfs_b, 4096);
  add_gelu_kernel<<<EB, 256, 0, stream>>>(x, R1, R3, E);
  transpose_vt_kernel<<<EB, 256, 0, stream>>>(R3, Yt, Nn);

  // ---- temporal branch ----
  gemm(Yt, w_in_t, b_in_t, QKT, R, 128, 64, 1, 0, 0, 0, 64, 128, 128, 30, 1, 0);
  tatt_kernel<<<(Nn * 16384 + 255) / 256, 256, 0, stream>>>(QKT, alphat_f, alphat_b, ATTT, Nn);
  for (int ds = 0; ds < 4; ++ds) {
    // z_{f/b}: per (n): Att^T(64x64) @ Yt(64x1600) scattered into Z256 cols dir*128 + c*2 + s
    gemm(ATTT + ds * 4096, Yt, nullptr, Z256 + (ds / 2) * 128 + (ds % 2),
         64, 1600, 64, Nn,
         16384, 102400, 409600,
         64, 1600, 6400,
         6, 2, 256);
  }
  gemm(Z256, w_out_t, b_out_t, R1, R, 64, 256, 1, 0, 0, 0, 256, 64, 64, 30, 1, 0);
  ln2_kernel<<<Nn * 64, 256, 0, stream>>>(R1, lnot_g, lnot_b, 1600);
  add_gelu_kernel<<<EB, 256, 0, stream>>>(Yt, R1, R2, E);
  gemm(R2, w_ff_t, b_ff_t, R1, R, 64, 64, 1, 0, 0, 0, 64, 64, 64, 30, 1, 0);
  ln2_kernel<<<Nn * 64, 256, 0, stream>>>(R1, lnft_g, lnft_b, 1600);
  add_gelu_kernel<<<EB, 256, 0, stream>>>(Yt, R1, R3, E);   // R3 = z (N,T,V,C)

  // ---- conv + batchnorm + output ----
  conv_kernel<<<EB, 256, 0, stream>>>(R3, conv_w, conv_b, R2, Nn);
  zero_kernel<<<1, 128, 0, stream>>>(STATS, 128);
  bnstats_kernel<<<512, 256, 0, stream>>>(R2, STATS, (long long)R);
  final_kernel<<<EB, 256, 0, stream>>>(R3, R2, STATS, bn_g, bn_b, out, Nn);
}
